// EditLMHF_12738873000399
// MI455X (gfx1250) — compile-verified, matrix operand order checked by reading, not run on hardware
//
#include <hip/hip_runtime.h>

// ---------------------------------------------------------------------------
// Problem constants (from reference): B=2, L=1024, D=768, V=50272
// G = L+1 = 1025, M = B*G = 2050 (padded to 2064 = 129*16), K3 = 3*D = 2304
// d_out layout (flat, f32): [0]=loss [1]=tok_loss [2]=idx_loss
//                           [3 .. 3+2050)          idx_logits (B,G)
//                           [3+2050 .. +100544)    tok_logits (B,V)
// ---------------------------------------------------------------------------
#define BB    2
#define LL    1024
#define DD    768
#define VV    50272
#define GG    1025
#define MM    2050
#define MPAD  2064          // 129 * 16
#define KK3   2304

typedef __attribute__((ext_vector_type(16))) __bf16        v16bf;
typedef __attribute__((ext_vector_type(8)))  float         v8f;
typedef __attribute__((ext_vector_type(4)))  unsigned int  v4u;

union FragBF {
    v16bf f;
    v4u   q[2];
};

__device__ __forceinline__ unsigned short f32_to_bf16_rne(float f) {
    unsigned int x = __float_as_uint(f);
    return (unsigned short)((x + 0x7FFFu + ((x >> 16) & 1u)) >> 16);
}

// ---------------------------------------------------------------------------
// W_triple f32 -> bf16 (RNE)
// ---------------------------------------------------------------------------
__global__ void cvt_bf16_kernel(const float* __restrict__ src,
                                unsigned short* __restrict__ dst, int n) {
    int i = blockIdx.x * blockDim.x + threadIdx.x;
    if (i < n) dst[i] = f32_to_bf16_rne(src[i]);
}

// ---------------------------------------------------------------------------
// Pack triple (MPAD x K3) in bf16, baking in the -1/0/+1 row shifts and
// zero padding. Row m = b*G + g; segment s of e selects h row g+s-1.
// Rows >= MM are zero. Removes ALL guards/divergence from the GEMM.
// grid = (K3/256, MPAD), block = 256
// ---------------------------------------------------------------------------
__global__ void pack_triple_kernel(const float* __restrict__ h,
                                   unsigned short* __restrict__ triple) {
    const int e = blockIdx.x * 256 + threadIdx.x;   // 0..2303
    const int m = blockIdx.y;                       // 0..2063
    unsigned short val = 0;
    if (m < MM) {
        const int b   = m / GG;
        const int g   = m - b * GG;
        const int seg = e / DD;                     // 0,1,2
        const int eo  = e - seg * DD;
        const int hrow = g + seg - 1;
        if (hrow >= 0 && hrow < LL)
            val = f32_to_bf16_rne(h[((size_t)b * LL + hrow) * DD + eo]);
    }
    triple[(size_t)m * KK3 + e] = val;
}

// ---------------------------------------------------------------------------
// GEMM1: gap[m, d] = sum_e triple[m, e] * W_triple[d, e]
//   v_wmma_f32_16x16x32_bf16, f32 accumulate. Each wave owns one 16-row
//   M-tile and TWO 16-col N-tiles (A fragment reused for 2 WMMAs).
//   Branch-free inner loop: 2 b128 loads for A, 2 for each B, 2 WMMA.
// grid = (129, 6), block = 128 (4 waves); nTiles: (by*4+wave)*2 + {0,1}
// ---------------------------------------------------------------------------
__global__ void gemm1_wmma_kernel(const unsigned short* __restrict__ triple,
                                  const unsigned short* __restrict__ wtbf,
                                  float* __restrict__ gap) {
    const int lane  = threadIdx.x & 31;
    const int wave  = threadIdx.x >> 5;
    const int mTile = blockIdx.x;                      // 0..128
    const int nt0   = (blockIdx.y * 4 + wave) * 2;     // 0,2,..,46

    const int sel = lane >> 4;    // 0: K{0..7,16..23}; 1: K{8..15,24..31}
    const int mi  = lane & 15;

    // Per-lane base pointers (A row = triple row; B col n = W_triple row d)
    const unsigned short* ap  = triple + ((size_t)(mTile * 16 + mi) * KK3 + sel * 8);
    const unsigned short* bp0 = wtbf   + ((size_t)(nt0 * 16 + mi) * KK3 + sel * 8);
    const unsigned short* bp1 = bp0 + (size_t)16 * KK3;

    v8f acc0 = {0.f, 0.f, 0.f, 0.f, 0.f, 0.f, 0.f, 0.f};
    v8f acc1 = {0.f, 0.f, 0.f, 0.f, 0.f, 0.f, 0.f, 0.f};

#pragma unroll 4
    for (int kc = 0; kc < KK3 / 32; ++kc) {            // 72 chunks of K=32
        const int off = kc * 32;
        FragBF a, b0, b1;
        a.q[0]  = *(const v4u*)(ap  + off);
        a.q[1]  = *(const v4u*)(ap  + off + 16);
        b0.q[0] = *(const v4u*)(bp0 + off);
        b0.q[1] = *(const v4u*)(bp0 + off + 16);
        b1.q[0] = *(const v4u*)(bp1 + off);
        b1.q[1] = *(const v4u*)(bp1 + off + 16);
        acc0 = __builtin_amdgcn_wmma_f32_16x16x32_bf16(
                   false, a.f, false, b0.f, (short)0, acc0, false, false);
        acc1 = __builtin_amdgcn_wmma_f32_16x16x32_bf16(
                   false, a.f, false, b1.f, (short)0, acc1, false, false);
    }

    // C/D layout: VGPR r -> M = (lane>=16 ? 8 : 0) + r, N = lane&15
    const int mh = sel * 8;
    float* grow = gap + (size_t)(mTile * 16 + mh) * DD;
#pragma unroll
    for (int r = 0; r < 8; ++r) {
        grow[(size_t)r * DD + nt0 * 16 + mi]        = acc0[r];
        grow[(size_t)r * DD + (nt0 + 1) * 16 + mi]  = acc1[r];
    }
}

// ---------------------------------------------------------------------------
// idx_logits[m] = dot(gap[m,:], w_index)   (2050 rows)
// ---------------------------------------------------------------------------
__global__ void idx_logits_kernel(const float* __restrict__ gap,
                                  const float* __restrict__ w_index,
                                  float* __restrict__ out_idx) {
    __shared__ float s[128];
    const int m = blockIdx.x;
    const float* gr = gap + (size_t)m * DD;
    float p = 0.f;
    for (int e = threadIdx.x; e < DD; e += 128) p += gr[e] * w_index[e];
    s[threadIdx.x] = p;
    __syncthreads();
    for (int o = 64; o > 0; o >>= 1) {
        if (threadIdx.x < o) s[threadIdx.x] += s[threadIdx.x + o];
        __syncthreads();
    }
    if (threadIdx.x == 0) out_idx[m] = s[0];
}

// ---------------------------------------------------------------------------
// tok_logits[b, v] = dot(gap[b, target_index[b], :], W_token[v, :])
//   Memory-bound: streams W_token (154 MB) once -> ~6.6 us @ 23.3 TB/s.
//   One wave per row, float4 coalesced loads, software prefetch 2 rows ahead.
// ---------------------------------------------------------------------------
__global__ void tok_logits_kernel(const float* __restrict__ gap,
                                  const float* __restrict__ W_token,
                                  const int* __restrict__ t_idx,
                                  float* __restrict__ out_tok) {
    __shared__ float gv[DD];
    const int b  = blockIdx.y;
    int tg = t_idx[b];
    if (tg < 0)   tg = 0;
    if (tg >= GG) tg = GG - 1;
    const float* grow = gap + ((size_t)b * GG + tg) * DD;
    for (int i = threadIdx.x; i < DD; i += blockDim.x) gv[i] = grow[i];
    __syncthreads();

    const int lane = threadIdx.x & 31;
    const int wave = threadIdx.x >> 5;
    const float4* gr = (const float4*)gv;

    for (int r = 0; r < 8; ++r) {
        const int v = blockIdx.x * 64 + wave * 8 + r;
        if (v < VV) {
            if (v + 2 < VV)   // global_prefetch_b8 two rows ahead
                __builtin_prefetch(W_token + (size_t)(v + 2) * DD + lane * 16, 0, 0);
            const float4* wr = (const float4*)(W_token + (size_t)v * DD);
            float sum = 0.f;
#pragma unroll
            for (int it = 0; it < DD / 128; ++it) {     // 6 float4 per lane
                float4 a  = wr[lane + it * 32];
                float4 gq = gr[lane + it * 32];
                sum += a.x * gq.x + a.y * gq.y + a.z * gq.z + a.w * gq.w;
            }
            for (int o = 16; o > 0; o >>= 1) sum += __shfl_down(sum, o, 32);
            if (lane == 0) out_tok[(size_t)b * VV + v] = sum;
        }
    }
}

// ---------------------------------------------------------------------------
// Final losses: single block, block-wide max/sum softmax reductions.
// ---------------------------------------------------------------------------
__device__ float blockReduceSum256(float v) {
    __shared__ float s[256];
    const int t = threadIdx.x;
    s[t] = v; __syncthreads();
    for (int o = 128; o > 0; o >>= 1) {
        if (t < o) s[t] += s[t + o];
        __syncthreads();
    }
    float r = s[0]; __syncthreads();
    return r;
}

__device__ float blockReduceMax256(float v) {
    __shared__ float s[256];
    const int t = threadIdx.x;
    s[t] = v; __syncthreads();
    for (int o = 128; o > 0; o >>= 1) {
        if (t < o) s[t] = fmaxf(s[t], s[t + o]);
        __syncthreads();
    }
    float r = s[0]; __syncthreads();
    return r;
}

__device__ float ce_row(const float* row, int n, int target) {
    float mx = -3.402823466e38f;
    for (int i = threadIdx.x; i < n; i += 256) mx = fmaxf(mx, row[i]);
    mx = blockReduceMax256(mx);
    float s = 0.f;
    for (int i = threadIdx.x; i < n; i += 256) s += __expf(row[i] - mx);
    s = blockReduceSum256(s);
    return -(row[target] - mx - __logf(s));
}

__global__ void loss_kernel(const int* __restrict__ t_idx,
                            const int* __restrict__ t_tok,
                            float* __restrict__ out) {
    const float* idx_logits = out + 3;
    const float* tok_logits = out + 3 + MM;

    float tok_loss = 0.f, idx_loss = 0.f;
    for (int b = 0; b < BB; ++b) {
        int tt = t_tok[b]; if (tt < 0) tt = 0; if (tt >= VV) tt = VV - 1;
        tok_loss += ce_row(tok_logits + (size_t)b * VV, VV, tt);
        int ti = t_idx[b]; if (ti < 0) ti = 0; if (ti >= GG) ti = GG - 1;
        idx_loss += ce_row(idx_logits + (size_t)b * GG, GG, ti);
    }
    tok_loss *= (1.f / BB);
    idx_loss *= (1.f / BB);
    if (threadIdx.x == 0) {
        out[0] = tok_loss + idx_loss;
        out[1] = tok_loss;
        out[2] = idx_loss;
    }
}

// ---------------------------------------------------------------------------
// Host-side launcher
// ---------------------------------------------------------------------------
extern "C" void kernel_launch(void* const* d_in, const int* in_sizes, int n_in,
                              void* d_out, int out_size, void* d_ws, size_t ws_size,
                              hipStream_t stream) {
    const float* h       = (const float*)d_in[0];   // (B, L, D)
    const float* W_trip  = (const float*)d_in[1];   // (D, 3D)
    const float* w_index = (const float*)d_in[2];   // (D,)
    const float* W_token = (const float*)d_in[3];   // (V, D)
    const int*   t_idx   = (const int*)d_in[4];     // (B,)
    const int*   t_tok   = (const int*)d_in[5];     // (B,)
    float*       out     = (float*)d_out;

    // Workspace layout (256B-aligned): wtbf | triple_bf | gap
    const size_t n_wt     = (size_t)DD * KK3;                // 1,769,472 bf16
    const size_t wt_bytes = (n_wt * 2 + 255) & ~(size_t)255;
    const size_t n_tr     = (size_t)MPAD * KK3;              // 4,755,456 bf16
    const size_t tr_bytes = (n_tr * 2 + 255) & ~(size_t)255;
    char* ws = (char*)d_ws;
    unsigned short* wtbf   = (unsigned short*)(ws);
    unsigned short* triple = (unsigned short*)(ws + wt_bytes);
    float*          gap    = (float*)(ws + wt_bytes + tr_bytes);  // MPAD x DD

    // 1) W_triple -> bf16 ; build padded bf16 triple with shifts baked in
    cvt_bf16_kernel<<<(int)((n_wt + 255) / 256), 256, 0, stream>>>(W_trip, wtbf, (int)n_wt);
    dim3 gp(KK3 / 256, MPAD);
    pack_triple_kernel<<<gp, 256, 0, stream>>>(h, triple);

    // 2) gap = triple @ W_triple^T via bf16 WMMA (branch-free, 2 N-tiles/wave)
    dim3 g1(129, 6);
    gemm1_wmma_kernel<<<g1, 128, 0, stream>>>(triple, wtbf, gap);

    // 3) idx_logits
    idx_logits_kernel<<<MM, 128, 0, stream>>>(gap, w_index, out + 3);

    // 4) tok_logits (only the target_index row per batch)
    dim3 g4((VV + 63) / 64, BB);
    tok_logits_kernel<<<g4, 256, 0, stream>>>(gap, W_token, t_idx, out + 3 + MM);

    // 5) cross-entropy losses
    loss_kernel<<<1, 256, 0, stream>>>(t_idx, t_tok, out);
}
